// DiffusionHead_35132832482009
// MI455X (gfx1250) — matrix operand order repaired and checked by reading, not compiled, not run on hardware
//
#include <hip/hip_runtime.h>
#include <hip/hip_bf16.h>
#include <math.h>

// ---------------------------------------------------------------------------
// CDNA5 (gfx1250) persistent diffusion-head kernel.
//   - whole T=32 scan in one kernel; x_t state lives in LDS (fp32) so GEMM
//     register pressure stays under 256 VGPRs (no spills, no vgpr-msb)
//   - layer1 folded: cond@W1c precomputed once (cond_proj in LDS),
//     temb@W1t+b1 precomputed per-step on device (sb table)
//   - all GEMMs via v_wmma_f32_16x16x32_bf16, f32 accum
//   - weights packed to bf16 transposed padded panels in d_ws (prep kernel),
//     staged to LDS each layer/step
// ---------------------------------------------------------------------------

typedef __bf16 bf16_t;
typedef __attribute__((ext_vector_type(16))) __bf16 v16bf;
typedef __attribute__((ext_vector_type(8)))  float  v8f;

union Frag16 { v16bf v; uint4 q[2]; };

#define T_STEPS  32
#define HID      256
#define CHUNK    64
#define ROWS     128            // rows per block
#define THREADS  256            // 8 wave32
#define HP       264            // bf16 pitch for 256-wide panels (528B rows)
#define XP       72             // bf16 pitch for 64-wide panels (144B rows)
#define SP       66             // fp32 pitch for x-state (264B rows)

// d_ws byte offsets (all 16B aligned)
#define OFF_SB   0                       // float [32][256]        32768 B
#define OFF_W1X  32768                   // bf16  [256][72]        36864 B
#define OFF_W1C  69632                   // bf16  [256][264]      135168 B
#define OFF_W2   204800                  // bf16  [256][264]      135168 B
#define OFF_W3   339968                  // bf16  [256][264]      135168 B
#define OFF_W4   475136                  // bf16  [64][264]        33792 B

// ---------------------------------------------------------------------------
__device__ __forceinline__ v8f wmma_bf16(v16bf a, v16bf b, v8f c) {
  return __builtin_amdgcn_wmma_f32_16x16x32_bf16(
      /*neg_a=*/false, a, /*neg_b=*/false, b,
      /*c_mod=*/(short)0, c, /*reuse_a=*/false, /*reuse_b=*/false);
}

__device__ __forceinline__ float silu_f(float v) {
  return v / (1.0f + expf(-v));
}

// ---- Threefry-2x32 (20 rounds) + Box-Muller ------------------------------
__device__ __forceinline__ unsigned rotl32(unsigned x, int r) {
  return (x << r) | (x >> (32 - r));
}
__device__ __forceinline__ void threefry2x32(unsigned k0, unsigned k1,
                                             unsigned c0, unsigned c1,
                                             unsigned& o0, unsigned& o1) {
  unsigned ks2 = k0 ^ k1 ^ 0x1BD11BDAu;
  unsigned x0 = c0 + k0, x1 = c1 + k1;
#define TF_RND(r) { x0 += x1; x1 = rotl32(x1, r); x1 ^= x0; }
  TF_RND(13) TF_RND(15) TF_RND(26) TF_RND(6)
  x0 += k1;  x1 += ks2 + 1u;
  TF_RND(17) TF_RND(29) TF_RND(16) TF_RND(24)
  x0 += ks2; x1 += k0 + 2u;
  TF_RND(13) TF_RND(15) TF_RND(26) TF_RND(6)
  x0 += k0;  x1 += k1 + 3u;
  TF_RND(17) TF_RND(29) TF_RND(16) TF_RND(24)
  x0 += k1;  x1 += ks2 + 4u;
  TF_RND(13) TF_RND(15) TF_RND(26) TF_RND(6)
  x0 += ks2; x1 += k0 + 5u;
#undef TF_RND
  o0 = x0; o1 = x1;
}
__device__ __forceinline__ float rand_normal(unsigned key, unsigned long long gid) {
  unsigned o0, o1;
  threefry2x32(key, 0x9E3779B9u, (unsigned)gid,
               (unsigned)(gid >> 32) ^ 0xCAFEF00Du, o0, o1);
  float u1 = ((float)(o0 >> 8) + 0.5f) * (1.0f / 16777216.0f);
  float u2 = ((float)(o1 >> 8) + 0.5f) * (1.0f / 16777216.0f);
  float r = sqrtf(-2.0f * logf(u1));
  return r * cosf(6.28318530718f * u2);
}

// ---------------------------------------------------------------------------
// Prep: pack W[K][N] (fp32 row-major) -> bf16 transposed padded panel [N][pitch]
__global__ void pack_wT(const float* __restrict__ src, bf16_t* __restrict__ dst,
                        int K, int N, int pitch) {
  int idx = blockIdx.x * blockDim.x + threadIdx.x;
  if (idx >= K * N) return;
  int k = idx % K, n = idx / K;
  dst[n * pitch + k] = (bf16_t)src[k * N + n];
}

// Prep: sb[s][n] = b1[n] + sum_j sin/cos(t(s)*f_j) * W1t  (t(s)=31-s)
__global__ void make_sb(const float* __restrict__ W1, const float* __restrict__ b1,
                        float* __restrict__ sb) {
  int idx = blockIdx.x * blockDim.x + threadIdx.x;
  if (idx >= T_STEPS * HID) return;
  int n = idx % HID, s = idx / HID;
  float t = (float)(31 - s);
  const float lf = 6.907755278982137f / 15.0f;   // ln(1000)/15
  float acc = b1[n];
  for (int j = 0; j < 16; ++j) {
    float f = expf(lf * (float)j);
    float a = t * f;
    acc += sinf(a) * W1[(64 + j) * HID + n];        // sin half of t_emb
    acc += cosf(a) * W1[(64 + 16 + j) * HID + n];   // cos half
  }
  sb[s * HID + n] = acc;
}

// ---------------------------------------------------------------------------
__global__ __launch_bounds__(THREADS, 1)
void diffusion_main(const float* __restrict__ cond,
                    const float* __restrict__ x_init,
                    const float* __restrict__ b2,
                    const float* __restrict__ b3,
                    const float* __restrict__ b4,
                    const float* __restrict__ sb,
                    const bf16_t* __restrict__ pW1x,
                    const bf16_t* __restrict__ pW1c,
                    const bf16_t* __restrict__ pW2,
                    const bf16_t* __restrict__ pW3,
                    const bf16_t* __restrict__ pW4,
                    float* __restrict__ out) {
  __shared__ bf16_t bufH[ROWS * HP];    // activations (and initial cond bf16)
  __shared__ bf16_t condp[ROWS * HP];   // cond @ W1c, per-row, step-invariant
  __shared__ bf16_t xbuf[ROWS * XP];    // x_t as bf16 for layer-1 A fragments
  __shared__ float  xs[ROWS * SP];      // x_t carried state, fp32
  __shared__ bf16_t wbuf[256 * HP];     // staged weight panel (current layer)

  const int tid  = threadIdx.x;
  const int wave = tid >> 5;
  const int lane = tid & 31;
  const int lr   = lane & 15;           // row (A) or col (B/C) within tile
  const int hi   = lane >> 4;           // lane half
  const int row0 = wave * 16;           // this wave's row tile within block
  const size_t grow0 = (size_t)blockIdx.x * ROWS;

  auto stage = [&](bf16_t* dst, const bf16_t* src, int n16) {
    const uint4* s = (const uint4*)src;
    uint4* d = (uint4*)dst;
    for (int i = tid; i < n16; i += THREADS) d[i] = s[i];
  };
  // A fragment 16x32 bf16: lane lr = row; halves hold K {0-7,16-23}/{8-15,24-31}
  auto loadA = [&](const bf16_t* buf, int pitch, int k0) -> v16bf {
    Frag16 f;
    const bf16_t* p = buf + (size_t)(row0 + lr) * pitch + k0 + hi * 8;
    f.q[0] = *(const uint4*)p;
    f.q[1] = *(const uint4*)(p + 16);
    return f.v;
  };
  // B fragment 32x16 bf16 from transposed panel W[n][k]: lane lr = col,
  // halves hold K {0-7,16-23}/{8-15,24-31}, same as A.
  auto loadB = [&](const bf16_t* w, int pitch, int n0, int k0) -> v16bf {
    Frag16 f;
    const bf16_t* p = w + (size_t)(n0 + lr) * pitch + k0 + hi * 8;
    f.q[0] = *(const uint4*)p;
    f.q[1] = *(const uint4*)(p + 16);
    return f.v;
  };

  v8f acc[16];

  // ---- one-time: cond -> bf16 LDS; cond_proj = cond @ W1c; x state init ---
  {
    const float* c0 = cond + grow0 * HID;
    for (int i = tid * 4; i < ROWS * HID; i += THREADS * 4) {
      int r = i >> 8, k = i & 255;
      float4 v = *(const float4*)(c0 + i);
      bf16_t* d = &bufH[r * HP + k];
      d[0] = (bf16_t)v.x; d[1] = (bf16_t)v.y;
      d[2] = (bf16_t)v.z; d[3] = (bf16_t)v.w;
    }
    stage(wbuf, pW1c, 256 * HP * 2 / 16);
  }
  __syncthreads();

  {
#pragma unroll
    for (int nt = 0; nt < 16; ++nt)
#pragma unroll
      for (int e = 0; e < 8; ++e) acc[nt][e] = 0.0f;
#pragma unroll 1
    for (int kt = 0; kt < 8; ++kt) {
      v16bf a = loadA(bufH, HP, kt * 32);
#pragma unroll
      for (int nt = 0; nt < 16; ++nt) {
        v16bf b = loadB(wbuf, HP, nt * 16, kt * 32);
        acc[nt] = wmma_bf16(a, b, acc[nt]);
      }
    }
#pragma unroll
    for (int nt = 0; nt < 16; ++nt)
#pragma unroll
      for (int e = 0; e < 8; ++e)
        condp[(row0 + e + hi * 8) * HP + nt * 16 + lr] = (bf16_t)acc[nt][e];
  }

  // x_init -> xs (fp32 state) + xbuf (bf16 A-panel); own rows only
#pragma unroll
  for (int nt = 0; nt < 4; ++nt)
#pragma unroll
    for (int e = 0; e < 8; ++e) {
      int r = row0 + e + hi * 8, c = nt * 16 + lr;
      float v = x_init[(grow0 + r) * CHUNK + c];
      xs[r * SP + c] = v;
      xbuf[r * XP + c] = (bf16_t)v;
    }

  // ---- diffusion scan -----------------------------------------------------
#pragma unroll 1
  for (int s = 0; s < T_STEPS; ++s) {
    const int t = 31 - s;
    float abar = 1.0f;
#pragma unroll 1
    for (int i = 0; i <= t; ++i) {
      float beta = 1e-4f + (1e-2f - 1e-4f) * (float)i * (1.0f / 31.0f);
      abar *= (1.0f - beta);
    }
    const float beta_t = 1e-4f + (1e-2f - 1e-4f) * (float)t * (1.0f / 31.0f);
    const float s1     = sqrtf(1.0f - abar);
    const float inv_sa = 1.0f / sqrtf(abar);
    const float sqa    = sqrtf(1.0f - beta_t);
    const float sqb    = sqrtf(beta_t);
    const unsigned nkey = 42u * 0x85EBCA6Bu + (unsigned)s;

    __syncthreads();                       // prev step done with wbuf/bufH
    stage(wbuf, pW1x, 256 * XP * 2 / 16);
    __syncthreads();

    // L1: h1 = silu(x @ W1x + cond_proj + sb[s])  (xbuf -> bufH)
    const float* sbs = sb + s * HID;
#pragma unroll
    for (int nt = 0; nt < 16; ++nt) {
      float bias = sbs[nt * 16 + lr];
#pragma unroll
      for (int e = 0; e < 8; ++e)
        acc[nt][e] = bias + (float)condp[(row0 + e + hi * 8) * HP + nt * 16 + lr];
    }
#pragma unroll 1
    for (int kt = 0; kt < 2; ++kt) {
      v16bf a = loadA(xbuf, XP, kt * 32);
#pragma unroll
      for (int nt = 0; nt < 16; ++nt) {
        v16bf b = loadB(wbuf, XP, nt * 16, kt * 32);
        acc[nt] = wmma_bf16(a, b, acc[nt]);
      }
    }
#pragma unroll
    for (int nt = 0; nt < 16; ++nt)
#pragma unroll
      for (int e = 0; e < 8; ++e)
        bufH[(row0 + e + hi * 8) * HP + nt * 16 + lr] = (bf16_t)silu_f(acc[nt][e]);
    __syncthreads();
    stage(wbuf, pW2, 256 * HP * 2 / 16);
    __syncthreads();

    // L2: h2 = silu(h1 @ W2 + b2)  (bufH in-place, per-wave rows)
#pragma unroll
    for (int nt = 0; nt < 16; ++nt) {
      float bias = b2[nt * 16 + lr];
#pragma unroll
      for (int e = 0; e < 8; ++e) acc[nt][e] = bias;
    }
#pragma unroll 1
    for (int kt = 0; kt < 8; ++kt) {
      v16bf a = loadA(bufH, HP, kt * 32);
#pragma unroll
      for (int nt = 0; nt < 16; ++nt) {
        v16bf b = loadB(wbuf, HP, nt * 16, kt * 32);
        acc[nt] = wmma_bf16(a, b, acc[nt]);
      }
    }
#pragma unroll
    for (int nt = 0; nt < 16; ++nt)
#pragma unroll
      for (int e = 0; e < 8; ++e)
        bufH[(row0 + e + hi * 8) * HP + nt * 16 + lr] = (bf16_t)silu_f(acc[nt][e]);
    __syncthreads();
    stage(wbuf, pW3, 256 * HP * 2 / 16);
    __syncthreads();

    // L3: h3 = silu(h2 @ W3 + b3)  (bufH in-place, per-wave rows)
#pragma unroll
    for (int nt = 0; nt < 16; ++nt) {
      float bias = b3[nt * 16 + lr];
#pragma unroll
      for (int e = 0; e < 8; ++e) acc[nt][e] = bias;
    }
#pragma unroll 1
    for (int kt = 0; kt < 8; ++kt) {
      v16bf a = loadA(bufH, HP, kt * 32);
#pragma unroll
      for (int nt = 0; nt < 16; ++nt) {
        v16bf b = loadB(wbuf, HP, nt * 16, kt * 32);
        acc[nt] = wmma_bf16(a, b, acc[nt]);
      }
    }
#pragma unroll
    for (int nt = 0; nt < 16; ++nt)
#pragma unroll
      for (int e = 0; e < 8; ++e)
        bufH[(row0 + e + hi * 8) * HP + nt * 16 + lr] = (bf16_t)silu_f(acc[nt][e]);
    __syncthreads();
    stage(wbuf, pW4, 64 * HP * 2 / 16);
    __syncthreads();

    // L4: eps = h3 @ W4 + b4   (N = 64 -> acc[0..3])
#pragma unroll
    for (int nt = 0; nt < 4; ++nt) {
      float bias = b4[nt * 16 + lr];
#pragma unroll
      for (int e = 0; e < 8; ++e) acc[nt][e] = bias;
    }
#pragma unroll 1
    for (int kt = 0; kt < 8; ++kt) {
      v16bf a = loadA(bufH, HP, kt * 32);
#pragma unroll
      for (int nt = 0; nt < 4; ++nt) {
        v16bf b = loadB(wbuf, HP, nt * 16, kt * 32);
        acc[nt] = wmma_bf16(a, b, acc[nt]);
      }
    }

    // x_{t-1} update: read/write fp32 state in LDS, refresh bf16 A-panel
#pragma unroll
    for (int nt = 0; nt < 4; ++nt)
#pragma unroll
      for (int e = 0; e < 8; ++e) {
        int r = row0 + e + hi * 8, c = nt * 16 + lr;
        float x0 = (xs[r * SP + c] - s1 * acc[nt][e]) * inv_sa;
        if (t > 0) {
          unsigned long long gid =
              (unsigned long long)(grow0 + r) * CHUNK + c;
          x0 = sqa * x0 + sqb * rand_normal(nkey, gid);
        }
        xs[r * SP + c] = x0;
        xbuf[r * XP + c] = (bf16_t)x0;
      }
  }

  // ---- write final x ------------------------------------------------------
#pragma unroll
  for (int nt = 0; nt < 4; ++nt)
#pragma unroll
    for (int e = 0; e < 8; ++e) {
      int r = row0 + e + hi * 8, c = nt * 16 + lr;
      out[(grow0 + r) * CHUNK + c] = xs[r * SP + c];
    }
}

// ---------------------------------------------------------------------------
extern "C" void kernel_launch(void* const* d_in, const int* in_sizes, int n_in,
                              void* d_out, int out_size, void* d_ws, size_t ws_size,
                              hipStream_t stream) {
  (void)in_sizes; (void)n_in; (void)out_size; (void)ws_size;
  const float* cond   = (const float*)d_in[0];
  const float* x_init = (const float*)d_in[1];
  const float* W1     = (const float*)d_in[2];
  const float* b1     = (const float*)d_in[3];
  const float* W2     = (const float*)d_in[4];
  const float* b2     = (const float*)d_in[5];
  const float* W3     = (const float*)d_in[6];
  const float* b3     = (const float*)d_in[7];
  const float* W4     = (const float*)d_in[8];
  const float* b4     = (const float*)d_in[9];

  char* ws = (char*)d_ws;
  float*  sbT  = (float*) (ws + OFF_SB);
  bf16_t* pW1x = (bf16_t*)(ws + OFF_W1X);
  bf16_t* pW1c = (bf16_t*)(ws + OFF_W1C);
  bf16_t* pW2  = (bf16_t*)(ws + OFF_W2);
  bf16_t* pW3  = (bf16_t*)(ws + OFF_W3);
  bf16_t* pW4  = (bf16_t*)(ws + OFF_W4);

  // weight packing (bf16, transposed, padded pitch)
  pack_wT<<<(64 * 256 + 255) / 256, 256, 0, stream>>>(W1,            pW1x,  64, 256, XP);
  pack_wT<<<(256 * 256 + 255) / 256, 256, 0, stream>>>(W1 + 96 * 256, pW1c, 256, 256, HP);
  pack_wT<<<(256 * 256 + 255) / 256, 256, 0, stream>>>(W2,            pW2,  256, 256, HP);
  pack_wT<<<(256 * 256 + 255) / 256, 256, 0, stream>>>(W3,            pW3,  256, 256, HP);
  pack_wT<<<(256 * 64  + 255) / 256, 256, 0, stream>>>(W4,            pW4,  256,  64, HP);
  make_sb<<<(T_STEPS * HID + 255) / 256, 256, 0, stream>>>(W1, b1, sbT);

  const int B = 65536;
  diffusion_main<<<B / ROWS, THREADS, 0, stream>>>(
      cond, x_init, b2, b3, b4, sbT, pW1x, pW1c, pW2, pW3, pW4, (float*)d_out);
}
